// Net_43490838839986
// MI455X (gfx1250) — matrix operand order repaired and checked by reading, not compiled, not run on hardware
//
#include <hip/hip_runtime.h>

// ---------------------------------------------------------------------------
// MI455X (gfx1250, wave32) GNN forward.  All dense 48x48 GEMMs run on
// v_wmma_f32_16x16x32_f16 (f16 in, f32 acc).  BatchNorm uses deterministic
// per-block partials (LDS ds_add_f32) + tiny reduce kernels.
// ---------------------------------------------------------------------------

typedef __attribute__((ext_vector_type(16))) _Float16 v16h;
typedef __attribute__((ext_vector_type(8)))  _Float16 v8h;
typedef __attribute__((ext_vector_type(8)))  float    v8f;

#define U      48
#define NN     25000
#define NPAD   25008          // N padded to 16-row tiles
#define EE     400000
#define DD     12
#define KS     16
#define WSEG   (DD*U*U)       // 27648 elems per [D,48,48] weight stack
#define BN_EPS 1e-5f

#define BLK     256
#define GRID_EW 512           // edge-side WMMA / partial kernels
#define GRID_NA 504           // elementwise node kernels (504*256 % 48 == 0)
#define GRID_NG 200           // node GEMM kernel

__device__ __forceinline__ float sigmoidf_(float x) { return 1.f / (1.f + __expf(-x)); }
__device__ __forceinline__ float siluf_(float x)    { return x * sigmoidf_(x); }

__device__ __forceinline__ v8f vzero8() {
  v8f z;
#pragma unroll
  for (int i = 0; i < 8; ++i) z[i] = 0.f;
  return z;
}

// A fragment: 16x32 f16 tile from fp32 rows (ISA 7.12.2: K = e + 8h + 8*[e>=8]).
// kc selects K window [32kc, 32kc+32); K >= 48 zero-padded.
__device__ __forceinline__ v16h load_a_f32(const float* __restrict__ tile, int lane, int kc) {
  const int m = lane & 15, h = lane >> 4;
  const float4* q = (const float4*)(tile + m * U + kc * 32 + 8 * h);
  v16h a;
  float4 u0 = q[0], u1 = q[1];
  a[0] = (_Float16)u0.x; a[1] = (_Float16)u0.y; a[2] = (_Float16)u0.z; a[3] = (_Float16)u0.w;
  a[4] = (_Float16)u1.x; a[5] = (_Float16)u1.y; a[6] = (_Float16)u1.z; a[7] = (_Float16)u1.w;
  if (kc == 0) {
    float4 u2 = q[4], u3 = q[5];   // +16 floats
    a[8]  = (_Float16)u2.x; a[9]  = (_Float16)u2.y; a[10] = (_Float16)u2.z; a[11] = (_Float16)u2.w;
    a[12] = (_Float16)u3.x; a[13] = (_Float16)u3.y; a[14] = (_Float16)u3.z; a[15] = (_Float16)u3.w;
  } else {
#pragma unroll
    for (int i = 0; i < 8; ++i) a[8 + i] = (_Float16)0.f;
  }
  return a;
}

// B fragment: 32x16 f16, B = W^T so column n of B is row (nbase+n) of W (K = e + 8h + 8*[e>=8]).
__device__ __forceinline__ v16h load_b_h(const _Float16* __restrict__ W, int nbase, int lane, int kc) {
  const int n = lane & 15, h = lane >> 4;
  v16h b;
  const v8h* q = (const v8h*)(W + (nbase + n) * U + kc * 32 + 8 * h);
  v8h b0 = q[0];
#pragma unroll
  for (int i = 0; i < 8; ++i) b[i] = b0[i];
  if (kc == 0) {
    v8h b1 = q[2];
#pragma unroll
    for (int i = 0; i < 8; ++i) b[8 + i] = b1[i];
  } else {
#pragma unroll
    for (int i = 0; i < 8; ++i) b[8 + i] = (_Float16)0.f;   // K = 48..63 -> zero pad
  }
  return b;
}

// acc[3] += (16x48 rows) @ W^T  (W is 48x48 row-major, f16)
__device__ __forceinline__ void gemm48(const v16h a0, const v16h a1,
                                       const _Float16* __restrict__ W, v8f acc[3], int lane) {
#pragma unroll
  for (int nt = 0; nt < 3; ++nt) {
    v16h b0 = load_b_h(W, nt * 16, lane, 0);
    v16h b1 = load_b_h(W, nt * 16, lane, 1);
    acc[nt] = __builtin_amdgcn_wmma_f32_16x16x32_f16(false, a0, false, b0, (short)0, acc[nt], false, false);
    acc[nt] = __builtin_amdgcn_wmma_f32_16x16x32_f16(false, a1, false, b1, (short)0, acc[nt], false, false);
  }
}

// --------------------------- weight fp32 -> fp16 ---------------------------
__global__ void k_convert(const float* __restrict__ vW1, const float* __restrict__ vW2,
                          const float* __restrict__ vW3, const float* __restrict__ vW4,
                          const float* __restrict__ eW,  const float* __restrict__ pW0,
                          const float* __restrict__ pW1, const float* __restrict__ zW0,
                          _Float16* __restrict__ out) {
  const int total = 5 * WSEG + 3 * U * U;
  const int stride = gridDim.x * blockDim.x;
  for (int i = blockIdx.x * blockDim.x + threadIdx.x; i < total; i += stride) {
    float v;
    if      (i < 1 * WSEG) v = vW1[i];
    else if (i < 2 * WSEG) v = vW2[i - 1 * WSEG];
    else if (i < 3 * WSEG) v = vW3[i - 2 * WSEG];
    else if (i < 4 * WSEG) v = vW4[i - 3 * WSEG];
    else if (i < 5 * WSEG) v = eW [i - 4 * WSEG];
    else if (i < 5 * WSEG + U * U)     v = pW0[i - 5 * WSEG];
    else if (i < 5 * WSEG + 2 * U * U) v = pW1[i - 5 * WSEG - U * U];
    else                               v = zW0[i - 5 * WSEG - 2 * U * U];
    out[i] = (_Float16)v;
  }
}

// ------------------------------- init h, w ---------------------------------
__global__ void k_init(const float* __restrict__ x, const float* __restrict__ ea,
                       const float* __restrict__ v0W, const float* __restrict__ v0b,
                       const float* __restrict__ e0W, const float* __restrict__ e0b,
                       float* __restrict__ h, float* __restrict__ w) {
  const int total = NPAD * U + EE * U;
  const int stride = gridDim.x * blockDim.x;
  for (int idx = blockIdx.x * blockDim.x + threadIdx.x; idx < total; idx += stride) {
    if (idx < NPAD * U) {
      const int i = idx / U, c = idx % U;
      float v = 0.f;
      if (i < NN) v = siluf_(x[i * 2] * v0W[c * 2] + x[i * 2 + 1] * v0W[c * 2 + 1] + v0b[c]);
      h[idx] = v;                          // pad rows stay zero forever
    } else {
      const int k = idx - NPAD * U;
      const int e = k / U, c = k % U;
      w[k] = siluf_(ea[e] * e0W[c] + e0b[c]);
    }
  }
}

// --------------------- node-side 4 GEMMs (h1..h4) --------------------------
__global__ void k_node_gemm4(const float* __restrict__ h,
                             const _Float16* __restrict__ W1, const _Float16* __restrict__ W2,
                             const _Float16* __restrict__ W3, const _Float16* __restrict__ W4,
                             const float* __restrict__ b1, const float* __restrict__ b2,
                             const float* __restrict__ b3, const float* __restrict__ b4,
                             float* __restrict__ hpre, float* __restrict__ h2o,
                             float* __restrict__ h3o,  float* __restrict__ h4o) {
  const int lane = threadIdx.x & 31;
  const int n = lane & 15, hh = lane >> 4;
  const int wave  = (blockIdx.x * blockDim.x + threadIdx.x) >> 5;
  const int nwav  = (gridDim.x * blockDim.x) >> 5;
  const _Float16* Ws[4] = {W1, W2, W3, W4};
  const float*    bs[4] = {b1, b2, b3, b4};
  for (int t = wave; t < NPAD / 16; t += nwav) {
    const float* tile = h + t * 16 * U;
    const v16h a0 = load_a_f32(tile, lane, 0);
    const v16h a1 = load_a_f32(tile, lane, 1);
    float* outs[4] = {hpre, h2o, h3o, h4o};
#pragma unroll
    for (int mi = 0; mi < 4; ++mi) {
      v8f acc[3]; acc[0] = vzero8(); acc[1] = vzero8(); acc[2] = vzero8();
      gemm48(a0, a1, Ws[mi], acc, lane);
#pragma unroll
      for (int nt = 0; nt < 3; ++nt) {
        const int c = nt * 16 + n;
        const float bb = bs[mi][c];
#pragma unroll
        for (int r = 0; r < 8; ++r)
          outs[mi][(t * 16 + r + 8 * hh) * U + c] = acc[nt][r] + bb;
      }
    }
  }
}

// ------ msg aggregate (deg==16, edges contiguous per src) + BN partials ----
__global__ void k_node_agg(const float* __restrict__ w, const float* __restrict__ h2,
                           const int* __restrict__ dst, float* __restrict__ hpre,
                           float* __restrict__ partials) {
  __shared__ float s_sum[U], s_sq[U];
  if (threadIdx.x < U) { s_sum[threadIdx.x] = 0.f; s_sq[threadIdx.x] = 0.f; }
  __syncthreads();
  const int stride = gridDim.x * blockDim.x;   // multiple of 48 by launch config
  float lsum = 0.f, lsq = 0.f; int lc = -1;
  for (int idx = blockIdx.x * blockDim.x + threadIdx.x; idx < NN * U; idx += stride) {
    const int i = idx / U, c = idx % U; lc = c;
    float a = 0.f;
#pragma unroll 4
    for (int j = 0; j < KS; ++j) {
      const int e = i * KS + j;
      a += sigmoidf_(w[e * U + c]) * h2[dst[e] * U + c];
    }
    const float pre = hpre[idx] + a * (1.f / 16.f);
    hpre[idx] = pre;
    lsum += pre; lsq += pre * pre;
  }
  if (lc >= 0) { atomicAdd(&s_sum[lc], lsum); atomicAdd(&s_sq[lc], lsq); }
  __syncthreads();
  if (threadIdx.x < U) {
    partials[blockIdx.x * 96 + threadIdx.x]     = s_sum[threadIdx.x];
    partials[blockIdx.x * 96 + U + threadIdx.x] = s_sq[threadIdx.x];
  }
}

// --------------------- BN stats finalize (mean, rstd) ----------------------
__global__ void k_bn_reduce(const float* __restrict__ partials, int nblocks,
                            float invCount, float* __restrict__ stats) {
  const int c = threadIdx.x;
  if (c >= U) return;
  float s = 0.f, q = 0.f;
  for (int b = 0; b < nblocks; ++b) { s += partials[b * 96 + c]; q += partials[b * 96 + U + c]; }
  const float m = s * invCount;
  float v = q * invCount - m * m;
  v = v < 0.f ? 0.f : v;
  stats[c]     = m;
  stats[U + c] = __frsqrt_rn(v + BN_EPS);
}

// --------------------- residual += silu(batchnorm(pre)) --------------------
__global__ void k_bn_apply(float* __restrict__ out, const float* __restrict__ pre,
                           const float* __restrict__ stats, const float* __restrict__ g,
                           const float* __restrict__ b, int rows) {
  const int stride = gridDim.x * blockDim.x;
  for (int idx = blockIdx.x * blockDim.x + threadIdx.x; idx < rows * U; idx += stride) {
    const int c = idx % U;
    const float xn = (pre[idx] - stats[c]) * stats[U + c] * g[c] + b[c];
    out[idx] += siluf_(xn);
  }
}

// -------- edge-side GEMM + gathers (src==tile) + BN partials ---------------
__global__ void k_edge_pre(const float* __restrict__ w, const _Float16* __restrict__ We,
                           const float* __restrict__ be, const float* __restrict__ h3,
                           const float* __restrict__ h4, const int* __restrict__ dst,
                           float* __restrict__ wpre, float* __restrict__ partials) {
  __shared__ float s_sum[U], s_sq[U];
  if (threadIdx.x < U) { s_sum[threadIdx.x] = 0.f; s_sq[threadIdx.x] = 0.f; }
  __syncthreads();
  const int lane = threadIdx.x & 31;
  const int n = lane & 15, hh = lane >> 4;
  const int wave = (blockIdx.x * blockDim.x + threadIdx.x) >> 5;
  const int nwav = (gridDim.x * blockDim.x) >> 5;
  float lsum[3] = {0.f, 0.f, 0.f}, lsq[3] = {0.f, 0.f, 0.f};
  for (int t = wave; t < EE / 16; t += nwav) {
    if (t + nwav < EE / 16) __builtin_prefetch(w + (t + nwav) * 16 * U, 0, 0);
    const v16h a0 = load_a_f32(w + t * 16 * U, lane, 0);
    const v16h a1 = load_a_f32(w + t * 16 * U, lane, 1);
    v8f acc[3]; acc[0] = vzero8(); acc[1] = vzero8(); acc[2] = vzero8();
    gemm48(a0, a1, We, acc, lane);
    int dd[8];
#pragma unroll
    for (int r = 0; r < 8; ++r) dd[r] = dst[t * 16 + 8 * hh + r];   // row m = r+8hh
#pragma unroll
    for (int nt = 0; nt < 3; ++nt) {
      const int c = nt * 16 + n;
      const float base = h3[t * U + c] + be[c];    // src of every edge in tile t is t
#pragma unroll
      for (int r = 0; r < 8; ++r) {
        const float v = acc[nt][r] + base + h4[dd[r] * U + c];
        wpre[(t * 16 + r + 8 * hh) * U + c] = v;
        lsum[nt] += v; lsq[nt] += v * v;
      }
    }
  }
#pragma unroll
  for (int nt = 0; nt < 3; ++nt) {
    atomicAdd(&s_sum[nt * 16 + n], lsum[nt]);
    atomicAdd(&s_sq [nt * 16 + n], lsq [nt]);
  }
  __syncthreads();
  if (threadIdx.x < U) {
    partials[blockIdx.x * 96 + threadIdx.x]     = s_sum[threadIdx.x];
    partials[blockIdx.x * 96 + U + threadIdx.x] = s_sq[threadIdx.x];
  }
}

// ------------------------------ head: p1 -----------------------------------
__global__ void k_gemm_silu(const float* __restrict__ in, const _Float16* __restrict__ W,
                            const float* __restrict__ bias, float* __restrict__ out, int ntiles) {
  const int lane = threadIdx.x & 31;
  const int n = lane & 15, hh = lane >> 4;
  const int wave = (blockIdx.x * blockDim.x + threadIdx.x) >> 5;
  const int nwav = (gridDim.x * blockDim.x) >> 5;
  for (int t = wave; t < ntiles; t += nwav) {
    const v16h a0 = load_a_f32(in + t * 16 * U, lane, 0);
    const v16h a1 = load_a_f32(in + t * 16 * U, lane, 1);
    v8f acc[3]; acc[0] = vzero8(); acc[1] = vzero8(); acc[2] = vzero8();
    gemm48(a0, a1, W, acc, lane);
#pragma unroll
    for (int nt = 0; nt < 3; ++nt) {
      const int c = nt * 16 + n;
      const float bb = bias[c];
#pragma unroll
      for (int r = 0; r < 8; ++r) {
        const float v = acc[nt][r] + bb;
        out[(t * 16 + r + 8 * hh) * U + c] = siluf_(v);
      }
    }
  }
}

// --------- head: p2 = silu(p1@pW1+b), score = p2 . pW2 + pb2 ---------------
__global__ void k_head_scores(const float* __restrict__ p1, const _Float16* __restrict__ W,
                              const float* __restrict__ b, const float* __restrict__ pW2,
                              const float* __restrict__ pb2, float* __restrict__ scores) {
  const int lane = threadIdx.x & 31;
  const int n = lane & 15, hh = lane >> 4;
  const int wave = (blockIdx.x * blockDim.x + threadIdx.x) >> 5;
  const int nwav = (gridDim.x * blockDim.x) >> 5;
  const float p2b = pb2[0];
  for (int t = wave; t < EE / 16; t += nwav) {
    const v16h a0 = load_a_f32(p1 + t * 16 * U, lane, 0);
    const v16h a1 = load_a_f32(p1 + t * 16 * U, lane, 1);
    v8f acc[3]; acc[0] = vzero8(); acc[1] = vzero8(); acc[2] = vzero8();
    gemm48(a0, a1, W, acc, lane);
    float tr[8];
#pragma unroll
    for (int r = 0; r < 8; ++r) tr[r] = 0.f;
#pragma unroll
    for (int nt = 0; nt < 3; ++nt) {
      const int c = nt * 16 + n;
      const float bb = b[c], wv = pW2[c];
#pragma unroll
      for (int r = 0; r < 8; ++r) tr[r] += siluf_(acc[nt][r] + bb) * wv;
    }
#pragma unroll
    for (int r = 0; r < 8; ++r) {          // butterfly within each 16-lane half
      float v = tr[r];
      for (int m = 1; m < 16; m <<= 1) v += __shfl_xor(v, m, 32);
      tr[r] = v;
    }
    if (n == 0) {
#pragma unroll
      for (int r = 0; r < 8; ++r) scores[t * 16 + 8 * hh + r] = tr[r] + p2b;
    }
  }
}

// ----------- head: logZ partials = sum_e relu(w@zW0+b) . zW1 ----------------
__global__ void k_head_z(const float* __restrict__ w, const _Float16* __restrict__ W,
                         const float* __restrict__ b, const float* __restrict__ zW1,
                         float* __restrict__ partials) {
  __shared__ float s_z;
  if (threadIdx.x == 0) s_z = 0.f;
  __syncthreads();
  const int lane = threadIdx.x & 31;
  const int n = lane & 15;
  const int wave = (blockIdx.x * blockDim.x + threadIdx.x) >> 5;
  const int nwav = (gridDim.x * blockDim.x) >> 5;
  float local = 0.f;
  for (int t = wave; t < EE / 16; t += nwav) {
    const v16h a0 = load_a_f32(w + t * 16 * U, lane, 0);
    const v16h a1 = load_a_f32(w + t * 16 * U, lane, 1);
    v8f acc[3]; acc[0] = vzero8(); acc[1] = vzero8(); acc[2] = vzero8();
    gemm48(a0, a1, W, acc, lane);
#pragma unroll
    for (int nt = 0; nt < 3; ++nt) {
      const int c = nt * 16 + n;
      const float bb = b[c], wv = zW1[c];
#pragma unroll
      for (int r = 0; r < 8; ++r) {
        const float v = acc[nt][r] + bb;
        local += (v > 0.f ? v : 0.f) * wv;
      }
    }
  }
  atomicAdd(&s_z, local);
  __syncthreads();
  if (threadIdx.x == 0) partials[blockIdx.x] = s_z;
}

__global__ void k_zfinal(const float* __restrict__ partials, int nb,
                         const float* __restrict__ zb1, float* __restrict__ out) {
  if (blockIdx.x == 0 && threadIdx.x == 0) {
    float s = 0.f;
    for (int bIt = 0; bIt < nb; ++bIt) s += partials[bIt];
    out[0] = s / (float)EE + zb1[0];
  }
}

// -------------------- per-node softmax over its 16 edges -------------------
__global__ void k_softmax(const float* __restrict__ scores, float* __restrict__ out) {
  const int i = blockIdx.x * blockDim.x + threadIdx.x;
  if (i >= NN) return;
  const float* s = scores + i * KS;
  float m = s[0];
#pragma unroll
  for (int j = 1; j < KS; ++j) m = fmaxf(m, s[j]);
  float ex[KS]; float sum = 0.f;
#pragma unroll
  for (int j = 0; j < KS; ++j) { ex[j] = __expf(s[j] - m); sum += ex[j]; }
  const float inv = 1.f / sum;
#pragma unroll
  for (int j = 0; j < KS; ++j) out[i * KS + j] = ex[j] * inv;
}

// ===========================================================================
extern "C" void kernel_launch(void* const* d_in, const int* in_sizes, int n_in,
                              void* d_out, int out_size, void* d_ws, size_t ws_size,
                              hipStream_t stream) {
  (void)in_sizes; (void)n_in; (void)out_size; (void)ws_size;
  const float* x       = (const float*)d_in[0];
  const float* ea      = (const float*)d_in[1];
  const int*   ei      = (const int*)  d_in[2];
  const float* v0W     = (const float*)d_in[3];
  const float* v0b     = (const float*)d_in[4];
  const float* vW1     = (const float*)d_in[5];
  const float* vW2     = (const float*)d_in[6];
  const float* vW3     = (const float*)d_in[7];
  const float* vW4     = (const float*)d_in[8];
  const float* vb1     = (const float*)d_in[9];
  const float* vb2     = (const float*)d_in[10];
  const float* vb3     = (const float*)d_in[11];
  const float* vb4     = (const float*)d_in[12];
  const float* vbn_g   = (const float*)d_in[13];
  const float* vbn_b   = (const float*)d_in[14];
  const float* e0W     = (const float*)d_in[15];
  const float* e0b     = (const float*)d_in[16];
  const float* eW      = (const float*)d_in[17];
  const float* eb      = (const float*)d_in[18];
  const float* ebn_g   = (const float*)d_in[19];
  const float* ebn_b   = (const float*)d_in[20];
  const float* pW0     = (const float*)d_in[21];
  const float* pb0     = (const float*)d_in[22];
  const float* pW1     = (const float*)d_in[23];
  const float* pb1     = (const float*)d_in[24];
  const float* pW2     = (const float*)d_in[25];
  const float* pb2     = (const float*)d_in[26];
  const float* zW0     = (const float*)d_in[27];
  const float* zb0     = (const float*)d_in[28];
  const float* zW1     = (const float*)d_in[29];
  const float* zb1     = (const float*)d_in[30];
  const int* dst = ei + EE;                  // edge_index row 1

  // ---- workspace carve-up (256B aligned) ----
  char* base = (char*)d_ws;
  size_t off = 0;
  auto carve = [&](size_t bytes) -> void* {
    void* p = base + off;
    off = (off + bytes + 255) & ~(size_t)255;
    return p;
  };
  float* h    = (float*)carve((size_t)NPAD * U * 4);
  float* hpre = (float*)carve((size_t)NPAD * U * 4);
  float* h2   = (float*)carve((size_t)NPAD * U * 4);
  float* h3   = (float*)carve((size_t)NPAD * U * 4);
  float* h4   = (float*)carve((size_t)NPAD * U * 4);
  float* w    = (float*)carve((size_t)EE * U * 4);
  float* wpre = (float*)carve((size_t)EE * U * 4);
  float* scr  = (float*)carve((size_t)EE * 4);
  _Float16* wh = (_Float16*)carve((size_t)(5 * WSEG + 3 * U * U) * 2);
  float* part  = (float*)carve((size_t)GRID_EW * 96 * 4);
  float* stats = (float*)carve(96 * 4);
  float* zpart = (float*)carve((size_t)GRID_EW * 4);

  const _Float16* vW1h = wh;
  const _Float16* vW2h = wh + 1 * WSEG;
  const _Float16* vW3h = wh + 2 * WSEG;
  const _Float16* vW4h = wh + 3 * WSEG;
  const _Float16* eWh  = wh + 4 * WSEG;
  const _Float16* pW0h = wh + 5 * WSEG;
  const _Float16* pW1h = wh + 5 * WSEG + U * U;
  const _Float16* zW0h = wh + 5 * WSEG + 2 * U * U;

  k_convert<<<64, BLK, 0, stream>>>(vW1, vW2, vW3, vW4, eW, pW0, pW1, zW0, wh);
  k_init<<<GRID_EW, BLK, 0, stream>>>(x, ea, v0W, v0b, e0W, e0b, h, w);

  for (int l = 0; l < DD; ++l) {
    const int wo = l * U * U, bo = l * U;
    k_node_gemm4<<<GRID_NG, BLK, 0, stream>>>(h,
        vW1h + wo, vW2h + wo, vW3h + wo, vW4h + wo,
        vb1 + bo, vb2 + bo, vb3 + bo, vb4 + bo,
        hpre, h2, h3, h4);
    k_node_agg<<<GRID_NA, BLK, 0, stream>>>(w, h2, dst, hpre, part);
    k_bn_reduce<<<1, 64, 0, stream>>>(part, GRID_NA, 1.f / (float)NN, stats);
    k_bn_apply<<<GRID_NA, BLK, 0, stream>>>(h, hpre, stats, vbn_g + bo, vbn_b + bo, NN);
    k_edge_pre<<<GRID_EW, BLK, 0, stream>>>(w, eWh + wo, eb + bo, h3, h4, dst, wpre, part);
    k_bn_reduce<<<1, 64, 0, stream>>>(part, GRID_EW, 1.f / (float)EE, stats);
    k_bn_apply<<<GRID_EW, BLK, 0, stream>>>(w, wpre, stats, ebn_g + bo, ebn_b + bo, EE);
  }

  // heads
  k_gemm_silu<<<GRID_EW, BLK, 0, stream>>>(w, pW0h, pb0, wpre, EE / 16);   // p1 -> wpre
  k_head_scores<<<GRID_EW, BLK, 0, stream>>>(wpre, pW1h, pb1, pW2, pb2, scr);
  k_head_z<<<GRID_EW, BLK, 0, stream>>>(w, zW0h, zb0, zW1, zpart);
  k_softmax<<<(NN + BLK - 1) / BLK, BLK, 0, stream>>>(scr, (float*)d_out);
  k_zfinal<<<1, 32, 0, stream>>>(zpart, GRID_EW, zb1, (float*)d_out + EE);
}
